// SpatialGGB_81338090652030
// MI455X (gfx1250) — compile-verified, matrix operand order checked by reading, not run on hardware
//
#include <hip/hip_runtime.h>
#include <hip/hip_bf16.h>

#define B_   4
#define HW_  4096
#define C_   256
#define M_   (B_ * HW_)   // 16384 rows across batches for projections

typedef __attribute__((ext_vector_type(16))) __bf16 v16bf;
typedef __attribute__((ext_vector_type(8)))  float  v8f;

union FragU {
    v16bf v;
    unsigned short us[16];
    unsigned int   u32[8];
};

__device__ __forceinline__ unsigned short f2bf(float f) {
    unsigned int u = __float_as_uint(f);
    unsigned int r = (u + 0x7FFFu + ((u >> 16) & 1u)) >> 16;  // RNE
    return (unsigned short)r;
}

__device__ __forceinline__ unsigned int pack2bf(float lo, float hi) {
    return (unsigned int)f2bf(lo) | ((unsigned int)f2bf(hi) << 16);
}

__device__ __forceinline__ v8f wmma_bf16(v16bf a, v16bf b, v8f c) {
    return __builtin_amdgcn_wmma_f32_16x16x32_bf16(
        /*neg_a=*/false, a, /*neg_b=*/false, b,
        /*c_mod=*/(short)0, c, /*reuse_a=*/false, /*reuse_b=*/false);
}

// ---- Fragment loader --------------------------------------------------------
// ISA 7.12.2 16-bit A layout (also serves as B when the source holds the
// transposed matrix row-major, i.e. B[k][n] = src[n][k]):
//   lane l: row = l&15 ; K base = 8*(l>>4)
//   VGPR v (pair 2v,2v+1): k = Kbase + 16*(v>>2) + 2*(v&3)  (k,k+1 contiguous)
// The 8 elements within each 16-K group are contiguous -> two b128 loads
// (global_load_b128 or ds_load_b128 depending on the address space).
__device__ __forceinline__ v16bf frag_kcontig_bf16(const unsigned short* src, int ld,
                                                   int row0, int k0, int lane) {
    FragU f;
    const unsigned short* p = src + (size_t)(row0 + (lane & 15)) * ld;
    int kb = k0 + ((lane >> 4) << 3);
#pragma unroll
    for (int v = 0; v < 8; ++v) {
        int k = kb + ((v >> 2) << 4) + ((v & 3) << 1);
        f.u32[v] = *(const unsigned int*)(p + k);
    }
    return f.v;
}

// ---- Conversion kernels (one-time, bandwidth-bound) ------------------------
__global__ __launch_bounds__(256) void k_cvt(const float* __restrict__ in,
                                             unsigned short* __restrict__ out, int n) {
    int idx = (blockIdx.x * 256 + threadIdx.x) * 8;
    if (idx < n) {
        float4 a = *(const float4*)(in + idx);
        float4 b = *(const float4*)(in + idx + 4);
        uint4 o;
        o.x = pack2bf(a.x, a.y);
        o.y = pack2bf(a.z, a.w);
        o.z = pack2bf(b.x, b.y);
        o.w = pack2bf(b.z, b.w);
        *(uint4*)(out + idx) = o;
    }
}

// W (C x C, row-major) -> Wt[n][k] = bf16(W[k][n])
__global__ __launch_bounds__(256) void k_cvtw(const float* __restrict__ W,
                                              unsigned short* __restrict__ Wt) {
    int idx = blockIdx.x * 256 + threadIdx.x;   // 65536 threads
    int n = idx >> 8, k = idx & 255;
    Wt[(size_t)n * C_ + k] = f2bf(W[(size_t)k * C_ + n]);
}

// ---- Kernel 1: projection GEMM  Y = bf16( Xb * Wt^T + bias ) ---------------
// grid: (M_/64, C_/128), block 256 (8 waves). Wave: 16 rows x 64 cols.
// TRANS=true writes per-batch transposed output Yt[b][c][i] with b128 stores.
template <bool TRANS>
__global__ __launch_bounds__(256) void k_proj(const unsigned short* __restrict__ Xb,
                                              const unsigned short* __restrict__ Wt,
                                              const float* __restrict__ bias,
                                              unsigned short* __restrict__ Y) {
    int lane = threadIdx.x & 31, wv = threadIdx.x >> 5;
    int row0 = blockIdx.x * 64 + (wv & 3) * 16;
    int col0 = blockIdx.y * 128 + (wv >> 2) * 64;

    v8f acc[4] = {{}, {}, {}, {}};
#pragma unroll 2
    for (int k0 = 0; k0 < C_; k0 += 32) {
        v16bf a = frag_kcontig_bf16(Xb, C_, row0, k0, lane);
#pragma unroll
        for (int t = 0; t < 4; ++t) {
            v16bf b = frag_kcontig_bf16(Wt, C_, col0 + t * 16, k0, lane);
            acc[t] = wmma_bf16(a, b, acc[t]);
        }
    }
    int half = lane >> 4, col = lane & 15;
    if (!TRANS) {
#pragma unroll
        for (int t = 0; t < 4; ++t) {
            int c = col0 + t * 16 + col;
            float bv = bias[c];
#pragma unroll
            for (int r = 0; r < 8; ++r) {
                int row = row0 + r + half * 8;
                Y[(size_t)row * C_ + c] = f2bf(acc[t][r] + bv);
            }
        }
    } else {
        int bb = row0 / HW_;                     // 64 | HW_: uniform per block
        int i0 = row0 + half * 8 - bb * HW_;     // 8 consecutive i per lane
#pragma unroll
        for (int t = 0; t < 4; ++t) {
            int c = col0 + t * 16 + col;
            float bv = bias[c];
            uint4 o;
            o.x = pack2bf(acc[t][0] + bv, acc[t][1] + bv);
            o.y = pack2bf(acc[t][2] + bv, acc[t][3] + bv);
            o.z = pack2bf(acc[t][4] + bv, acc[t][5] + bv);
            o.w = pack2bf(acc[t][6] + bv, acc[t][7] + bv);
            *(uint4*)&Y[((size_t)bb * C_ + c) * HW_ + i0] = o;
        }
    }
}

// Dual-matrix logit step from LDS-staged j-tiles:
//   S += A_s x B(X1tile), T += A_t x B(G1tile); K = 256.
// B fragments come from LDS (ds_load_b128 pairs, short latency).
__device__ __forceinline__ void logit_tile_lds(const v16bf* a_s, const v16bf* a_t,
                                               const unsigned short* X1t,
                                               const unsigned short* G1t,
                                               int lane, v8f& s, v8f& t) {
#pragma unroll
    for (int kk = 0; kk < 8; ++kk) {
        v16bf bs = frag_kcontig_bf16(X1t, C_, 0, kk * 32, lane);
        s = wmma_bf16(a_s[kk], bs, s);
        v16bf bt = frag_kcontig_bf16(G1t, C_, 0, kk * 32, lane);
        t = wmma_bf16(a_t[kk], bt, t);
    }
}

// Shared LDS double-buffer machinery for the stats kernels.
// Per block (8 waves, each with its own 16-row i-tile) the j-stream of
// xw1/gw1 is identical -> stage each 16-j tile (8KB + 8KB) cooperatively.
#define STATS_PRELOAD_A()                                                     \
    v16bf a_s[8], a_t[8];                                                     \
    _Pragma("unroll")                                                         \
    for (int kk = 0; kk < 8; ++kk) {                                          \
        a_s[kk] = frag_kcontig_bf16(X2, C_, i0, kk * 32, lane);               \
        a_t[kk] = frag_kcontig_bf16(G2, C_, i0, kk * 32, lane);               \
    }

#define STATS_PRIME_LDS()                                                     \
    {                                                                         \
        const uint4* gX = (const uint4*)X1;                                   \
        const uint4* gG = (const uint4*)G1;                                   \
        uint4 px0 = gX[tid], px1 = gX[tid + 256];                             \
        uint4 pg0 = gG[tid], pg1 = gG[tid + 256];                             \
        ((uint4*)sJ[0][0])[tid] = px0; ((uint4*)sJ[0][0])[tid + 256] = px1;   \
        ((uint4*)sJ[0][1])[tid] = pg0; ((uint4*)sJ[0][1])[tid + 256] = pg1;   \
    }                                                                         \
    __syncthreads();

#define STATS_LOAD_NEXT()                                                     \
    uint4 rx0, rx1, rg0, rg1;                                                 \
    bool hasNext = (j0 + 16) < HW_;                                           \
    if (hasNext) {                                                            \
        const uint4* gX = (const uint4*)(X1 + (size_t)(j0 + 16) * C_);        \
        const uint4* gG = (const uint4*)(G1 + (size_t)(j0 + 16) * C_);        \
        rx0 = gX[tid]; rx1 = gX[tid + 256];                                   \
        rg0 = gG[tid]; rg1 = gG[tid + 256];                                   \
    }

#define STATS_STORE_NEXT()                                                    \
    if (hasNext) {                                                            \
        unsigned short* dX = sJ[cur ^ 1][0];                                  \
        unsigned short* dG = sJ[cur ^ 1][1];                                  \
        ((uint4*)dX)[tid] = rx0; ((uint4*)dX)[tid + 256] = rx1;               \
        ((uint4*)dG)[tid] = rg0; ((uint4*)dG)[tid + 256] = rg1;               \
    }                                                                         \
    __syncthreads();                                                          \
    cur ^= 1;

// ---- Kernel 2: row stats pass 1: lx=LSE(s), lg=LSE(t), m=exp(max(s+t)-lx-lg)
__global__ __launch_bounds__(256) void k_stats1(const unsigned short* __restrict__ xw2,
                                                const unsigned short* __restrict__ xw1,
                                                const unsigned short* __restrict__ gw2,
                                                const unsigned short* __restrict__ gw1,
                                                float* __restrict__ lx,
                                                float* __restrict__ lg,
                                                float* __restrict__ m) {
    __shared__ __align__(16) unsigned short sJ[2][2][16 * C_];  // 32 KB

    int tid = threadIdx.x;
    int lane = tid & 31;
    int gwv = (blockIdx.x * 256 + tid) >> 5;
    int b = gwv >> 8;
    int i0 = (gwv & 255) * 16;
    size_t boff = (size_t)b * HW_ * C_;
    const unsigned short* X2 = xw2 + boff;
    const unsigned short* X1 = xw1 + boff;
    const unsigned short* G2 = gw2 + boff;
    const unsigned short* G1 = gw1 + boff;

    STATS_PRELOAD_A();

    float mxs[8], zs[8], mxt[8], zt[8], mst[8];
#pragma unroll
    for (int r = 0; r < 8; ++r) {
        mxs[r] = -1e30f; zs[r] = 0.f; mxt[r] = -1e30f; zt[r] = 0.f; mst[r] = -1e30f;
    }

    STATS_PRIME_LDS();
    int cur = 0;
    for (int j0 = 0; j0 < HW_; j0 += 16) {
        STATS_LOAD_NEXT();           // issue global loads, no wait yet
        v8f s = {}, t = {};
        logit_tile_lds(a_s, a_t, sJ[cur][0], sJ[cur][1], lane, s, t);
#pragma unroll
        for (int r = 0; r < 8; ++r) {
            float sv = s[r], tv = t[r];
            float nm = fmaxf(mxs[r], sv);
            zs[r] = zs[r] * __expf(mxs[r] - nm) + __expf(sv - nm);
            mxs[r] = nm;
            nm = fmaxf(mxt[r], tv);
            zt[r] = zt[r] * __expf(mxt[r] - nm) + __expf(tv - nm);
            mxt[r] = nm;
            mst[r] = fmaxf(mst[r], sv + tv);
        }
        STATS_STORE_NEXT();          // waits land here, behind a compute phase
    }
    // reduce across the 16 lanes of each half (each row lives in one half)
#pragma unroll
    for (int off = 1; off < 16; off <<= 1) {
#pragma unroll
        for (int r = 0; r < 8; ++r) {
            float om = __shfl_xor(mxs[r], off);
            float oz = __shfl_xor(zs[r], off);
            float nm = fmaxf(mxs[r], om);
            zs[r] = zs[r] * __expf(mxs[r] - nm) + oz * __expf(om - nm);
            mxs[r] = nm;
            om = __shfl_xor(mxt[r], off);
            oz = __shfl_xor(zt[r], off);
            nm = fmaxf(mxt[r], om);
            zt[r] = zt[r] * __expf(mxt[r] - nm) + oz * __expf(om - nm);
            mxt[r] = nm;
            mst[r] = fmaxf(mst[r], __shfl_xor(mst[r], off));
        }
    }
    if ((lane & 15) == 0) {
        int half = lane >> 4;
#pragma unroll
        for (int r = 0; r < 8; ++r) {
            int i = b * HW_ + i0 + half * 8 + r;
            float lxv = mxs[r] + __logf(zs[r]);
            float lgv = mxt[r] + __logf(zt[r]);
            lx[i] = lxv;
            lg[i] = lgv;
            m[i] = __expf(mst[r] - lxv - lgv);   // max_j p_ij
        }
    }
}

// ---- Kernel 3: row stats pass 2: rZ_i = 1 / sum_j exp(p_ij - m_i) ----------
__global__ __launch_bounds__(256) void k_stats2(const unsigned short* __restrict__ xw2,
                                                const unsigned short* __restrict__ xw1,
                                                const unsigned short* __restrict__ gw2,
                                                const unsigned short* __restrict__ gw1,
                                                const float* __restrict__ lx,
                                                const float* __restrict__ lg,
                                                const float* __restrict__ m,
                                                float* __restrict__ rZ) {
    __shared__ __align__(16) unsigned short sJ[2][2][16 * C_];  // 32 KB

    int tid = threadIdx.x;
    int lane = tid & 31;
    int gwv = (blockIdx.x * 256 + tid) >> 5;
    int b = gwv >> 8;
    int i0 = (gwv & 255) * 16;
    size_t boff = (size_t)b * HW_ * C_;
    const unsigned short* X2 = xw2 + boff;
    const unsigned short* X1 = xw1 + boff;
    const unsigned short* G2 = gw2 + boff;
    const unsigned short* G1 = gw1 + boff;

    STATS_PRELOAD_A();

    int half = lane >> 4;
    float lxr[8], lgr[8], mr[8], zz[8];
#pragma unroll
    for (int r = 0; r < 8; ++r) {
        int i = b * HW_ + i0 + half * 8 + r;
        lxr[r] = lx[i]; lgr[r] = lg[i]; mr[r] = m[i]; zz[r] = 0.f;
    }

    STATS_PRIME_LDS();
    int cur = 0;
    for (int j0 = 0; j0 < HW_; j0 += 16) {
        STATS_LOAD_NEXT();
        v8f s = {}, t = {};
        logit_tile_lds(a_s, a_t, sJ[cur][0], sJ[cur][1], lane, s, t);
#pragma unroll
        for (int r = 0; r < 8; ++r) {
            float p = __expf((s[r] - lxr[r]) + (t[r] - lgr[r]));
            zz[r] += __expf(p - mr[r]);
        }
        STATS_STORE_NEXT();
    }
#pragma unroll
    for (int off = 1; off < 16; off <<= 1)
#pragma unroll
        for (int r = 0; r < 8; ++r) zz[r] += __shfl_xor(zz[r], off);

    if ((lane & 15) == 0) {
#pragma unroll
        for (int r = 0; r < 8; ++r)
            rZ[b * HW_ + i0 + half * 8 + r] = 1.0f / zz[r];
    }
}

// ---- Kernel 4: out[b,j,c] = x[b,j,c] + sum_i a_ij * xw3[i,c] ---------------
// Block = one 16-column j-tile of one batch; i advances in 64-row chunks.
// Waves 0..3 recompute the four 16-row S sub-tiles, waves 4..7 the T sub-tiles
// (all 8 waves WMMA-active). All 256 threads form a_ij (bf16, transposed) in
// LDS, then all 8 waves run the K=64 out-GEMM for their 32-channel slice.
// xw3 is provided pre-transposed (xw3T[b][c][i]) so B fragments are K-contig.
__global__ __launch_bounds__(256) void k_out(const unsigned short* __restrict__ xw2,
                                             const unsigned short* __restrict__ xw1,
                                             const unsigned short* __restrict__ gw2,
                                             const unsigned short* __restrict__ gw1,
                                             const unsigned short* __restrict__ xw3T,
                                             const float* __restrict__ lx,
                                             const float* __restrict__ lg,
                                             const float* __restrict__ m,
                                             const float* __restrict__ rZ,
                                             const float* __restrict__ Xin,
                                             float* __restrict__ out) {
    __shared__ __align__(16) float sS[64][16];
    __shared__ __align__(16) float sT[64][16];
    __shared__ __align__(16) unsigned short sA[16][64];   // a^T[j][i] as bf16

    int lane = threadIdx.x & 31, wv = threadIdx.x >> 5;
    int b = blockIdx.x >> 8;
    int j0 = (blockIdx.x & 255) * 16;
    size_t boff = (size_t)b * HW_ * C_;
    const unsigned short* X2 = xw2 + boff;
    const unsigned short* X1 = xw1 + boff;
    const unsigned short* G2 = gw2 + boff;
    const unsigned short* G1 = gw1 + boff;
    const unsigned short* X3T = xw3T + boff;   // [c][i], ld = HW_
    const float* lxb = lx + b * HW_;
    const float* lgb = lg + b * HW_;
    const float* mb  = m  + b * HW_;
    const float* Zb  = rZ + b * HW_;

    int c0 = wv * 32;
    v8f acc0 = {}, acc1 = {};

    // j-side logit operand is loop-invariant; S waves use X1, T waves use G1.
    const unsigned short* Pj = (wv < 4) ? X1 : G1;
    const unsigned short* Pi = (wv < 4) ? X2 : G2;
    v16bf bj[8];
#pragma unroll
    for (int kk = 0; kk < 8; ++kk)
        bj[kk] = frag_kcontig_bf16(Pj, C_, j0, kk * 32, lane);

    for (int i0 = 0; i0 < HW_; i0 += 64) {
        // ---- logit recompute: every wave produces one 16x16 S or T sub-tile
        {
            int ir = i0 + (wv & 3) * 16;
            if (i0 + 64 < HW_)
                __builtin_prefetch(Pi + (size_t)(ir + 64 + (lane & 15)) * C_, 0, 1);
            v8f acc = {};
            v16bf bb[4];
#pragma unroll
            for (int p = 0; p < 2; ++p)
                bb[p] = frag_kcontig_bf16(Pi, C_, ir, p * 32, lane);
#pragma unroll
            for (int kk = 0; kk < 8; ++kk) {
                if (kk + 2 < 8)
                    bb[(kk + 2) & 3] = frag_kcontig_bf16(Pi, C_, ir, (kk + 2) * 32, lane);
                acc = wmma_bf16(bb[kk & 3], bj[kk], acc);
            }
            int half = lane >> 4, col = lane & 15;
            float (*dst)[16] = (wv < 4) ? sS : sT;
#pragma unroll
            for (int r = 0; r < 8; ++r)
                dst[(wv & 3) * 16 + r + half * 8][col] = acc[r];
        }
        __syncthreads();

        // ---- gate + double softmax -> a_ij (bf16, transposed for A-frag)
        for (int e = threadIdx.x; e < 1024; e += 256) {
            int il = e >> 4, jl = e & 15;
            int i = i0 + il;
            float p = __expf((sS[il][jl] - lxb[i]) + (sT[il][jl] - lgb[i]));
            float a = __expf(p - mb[i]) * Zb[i];
            sA[jl][il] = f2bf(a);
        }
        __syncthreads();

        // ---- out-GEMM: K = 64 over i, per-wave 32-channel slice
        v16bf a0 = frag_kcontig_bf16(&sA[0][0], 64, 0, 0, lane);
        v16bf a1 = frag_kcontig_bf16(&sA[0][0], 64, 0, 32, lane);
        v16bf b00 = frag_kcontig_bf16(X3T, HW_, c0,      i0,      lane);
        v16bf b01 = frag_kcontig_bf16(X3T, HW_, c0,      i0 + 32, lane);
        v16bf b10 = frag_kcontig_bf16(X3T, HW_, c0 + 16, i0,      lane);
        v16bf b11 = frag_kcontig_bf16(X3T, HW_, c0 + 16, i0 + 32, lane);
        acc0 = wmma_bf16(a0, b00, acc0);
        acc0 = wmma_bf16(a1, b01, acc0);
        acc1 = wmma_bf16(a0, b10, acc1);
        acc1 = wmma_bf16(a1, b11, acc1);
        __syncthreads();   // protect LDS before next chunk overwrites it
    }

    int half = lane >> 4, col = lane & 15;
#pragma unroll
    for (int r = 0; r < 8; ++r) {
        int j = j0 + r + half * 8;
        size_t base = ((size_t)b * HW_ + j) * C_;
        int cA = c0 + col, cB = c0 + 16 + col;
        out[base + cA] = Xin[base + cA] + acc0[r];
        out[base + cB] = Xin[base + cB] + acc1[r];
    }
}

// ---- Host launch ------------------------------------------------------------
extern "C" void kernel_launch(void* const* d_in, const int* in_sizes, int n_in,
                              void* d_out, int out_size, void* d_ws, size_t ws_size,
                              hipStream_t stream) {
    const float* x   = (const float*)d_in[0];
    const float* g   = (const float*)d_in[1];
    const float* Wx1 = (const float*)d_in[2];
    const float* bx1 = (const float*)d_in[3];
    const float* Wx2 = (const float*)d_in[4];
    const float* bx2 = (const float*)d_in[5];
    const float* Wx3 = (const float*)d_in[6];
    const float* bx3 = (const float*)d_in[7];
    const float* Wg1 = (const float*)d_in[8];
    const float* bg1 = (const float*)d_in[9];
    const float* Wg2 = (const float*)d_in[10];
    const float* bg2 = (const float*)d_in[11];
    float* outp = (float*)d_out;

    const size_t nProj = (size_t)M_ * C_;          // 4,194,304 elems
    const size_t nW    = (size_t)C_ * C_;          // 65,536 elems
    char* ws = (char*)d_ws;
    unsigned short* xb   = (unsigned short*)(ws + 0 * nProj * 2);
    unsigned short* gb   = (unsigned short*)(ws + 1 * nProj * 2);
    unsigned short* xw1b = (unsigned short*)(ws + 2 * nProj * 2);
    unsigned short* xw2b = (unsigned short*)(ws + 3 * nProj * 2);
    unsigned short* gw1b = (unsigned short*)(ws + 4 * nProj * 2);
    unsigned short* gw2b = (unsigned short*)(ws + 5 * nProj * 2);
    unsigned short* x3T  = (unsigned short*)(ws + 6 * nProj * 2);
    unsigned short* wt   = (unsigned short*)(ws + 7 * nProj * 2);
    unsigned short* Wt1  = wt + 0 * nW;
    unsigned short* Wt2  = wt + 1 * nW;
    unsigned short* Wt3  = wt + 2 * nW;
    unsigned short* Wtg1 = wt + 3 * nW;
    unsigned short* Wtg2 = wt + 4 * nW;
    float* lx = (float*)(wt + 5 * nW);
    float* lg = lx + M_;
    float* mS = lg + M_;
    float* ZS = mS + M_;

    // one-time bf16 materialization (bandwidth-bound)
    int cvtBlocks = (int)(nProj / (256 * 8));
    k_cvt<<<cvtBlocks, 256, 0, stream>>>(x, xb, (int)nProj);
    k_cvt<<<cvtBlocks, 256, 0, stream>>>(g, gb, (int)nProj);
    k_cvtw<<<256, 256, 0, stream>>>(Wx1, Wt1);
    k_cvtw<<<256, 256, 0, stream>>>(Wx2, Wt2);
    k_cvtw<<<256, 256, 0, stream>>>(Wx3, Wt3);
    k_cvtw<<<256, 256, 0, stream>>>(Wg1, Wtg1);
    k_cvtw<<<256, 256, 0, stream>>>(Wg2, Wtg2);

    dim3 pg(M_ / 64, C_ / 128);
    k_proj<false><<<pg, 256, 0, stream>>>(xb, Wt1, bx1, xw1b);
    k_proj<false><<<pg, 256, 0, stream>>>(xb, Wt2, bx2, xw2b);
    k_proj<true ><<<pg, 256, 0, stream>>>(xb, Wt3, bx3, x3T);
    k_proj<false><<<pg, 256, 0, stream>>>(gb, Wtg1, bg1, gw1b);
    k_proj<false><<<pg, 256, 0, stream>>>(gb, Wtg2, bg2, gw2b);

    // one wave per 16-row i-tile: B*HW/16 = 1024 waves -> 128 blocks of 8 waves
    k_stats1<<<128, 256, 0, stream>>>(xw2b, xw1b, gw2b, gw1b, lx, lg, mS);
    k_stats2<<<128, 256, 0, stream>>>(xw2b, xw1b, gw2b, gw1b, lx, lg, mS, ZS);

    // one block per 16-column j-tile per batch: 4*256 = 1024 blocks
    k_out<<<B_ * 256, 256, 0, stream>>>(xw2b, xw1b, gw2b, gw1b, x3T,
                                        lx, lg, mS, ZS, x, outp);
}
